// SelfAttention_76587856822477
// MI455X (gfx1250) — matrix operand order repaired and checked by reading, not compiled, not run on hardware
//
#include <hip/hip_runtime.h>
#include <math.h>

// Problem constants (from reference): B=2, L=4096, E=512, H=8, D=64
#define B_ 2
#define L_ 4096
#define E_ 512
#define H_ 8
#define D_ 64

// softmax scale folded with log2(e) so we can use exp2 (the HW transcendental)
#define CSC (0.044194173824159216f * 1.4426950408889634f)

typedef __attribute__((ext_vector_type(16))) _Float16 v16h;
typedef __attribute__((ext_vector_type(8)))  float    v8f;
typedef __attribute__((ext_vector_type(4)))  unsigned int u32x4;

union FragA {          // one 16x32 (A) or 32x16 (B) f16 wave fragment: 8 VGPRs
    v16h  h;
    u32x4 u[2];
};

static __device__ __forceinline__ v8f wmma16(const FragA& a, const FragA& b, v8f c) {
    // D = A(16x32 f16) x B(32x16 f16) + C(16x16 f32)
    return __builtin_amdgcn_wmma_f32_16x16x32_f16(
        /*neg_a=*/false, a.h, /*neg_b=*/false, b.h,
        /*c_mod=*/(short)0, c, /*reuse_a=*/false, /*reuse_b=*/false);
}

// --- CDNA5 async global->LDS copy (ASYNCcnt-tracked, bypasses VGPRs) -------
static __device__ __forceinline__ void async_copy16(unsigned ldsByteAddr,
                                                    const void* gaddr) {
    asm volatile("global_load_async_to_lds_b128 %0, %1, off"
                 :: "v"(ldsByteAddr),
                    "v"((unsigned long long)(size_t)gaddr)
                 : "memory");
}

#if __has_builtin(__builtin_amdgcn_s_wait_asynccnt)
#define S_WAIT_ASYNCCNT(n) __builtin_amdgcn_s_wait_asynccnt(n)
#else
#define S_WAIT_ASYNCCNT(n) asm volatile("s_wait_asynccnt " #n ::: "memory")
#endif

// --- 16-lane row reductions in the VALU pipe via DPP16 (no LDS traffic) ----
// xor1 = quad_perm[1,0,3,2]=0xB1, xor2 = quad_perm[2,3,0,1]=0x4E,
// then row_half_mirror (0x141) ~ xor4, row_mirror (0x140) ~ xor8.
template <int CTRL>
static __device__ __forceinline__ float dpp_maxstep(float x) {
    int v = __builtin_amdgcn_update_dpp(0, __float_as_int(x), CTRL, 0xf, 0xf, true);
    return fmaxf(x, __int_as_float(v));
}
template <int CTRL>
static __device__ __forceinline__ float dpp_addstep(float x) {
    int v = __builtin_amdgcn_update_dpp(0, __float_as_int(x), CTRL, 0xf, 0xf, true);
    return x + __int_as_float(v);
}
static __device__ __forceinline__ float row16_max(float x) {
    x = dpp_maxstep<0xB1>(x);
    x = dpp_maxstep<0x4E>(x);
    x = dpp_maxstep<0x141>(x);
    x = dpp_maxstep<0x140>(x);
    return x;
}
static __device__ __forceinline__ float row16_sum(float x) {
    x = dpp_addstep<0xB1>(x);
    x = dpp_addstep<0x4E>(x);
    x = dpp_addstep<0x141>(x);
    x = dpp_addstep<0x140>(x);
    return x;
}

// ---------------------------------------------------------------------------
// Prep 1: values (f32) -> Vh (f16, row major [b][l][e], used as both Q and V)
//         and Vt (f16, [b][h][d][l]) so PV B-tiles are contiguous in l.
// ---------------------------------------------------------------------------
__global__ __launch_bounds__(256) void prep_vals_kernel(
    const float* __restrict__ values, _Float16* __restrict__ Vh,
    _Float16* __restrict__ Vt) {
    int idx = blockIdx.x * 256 + threadIdx.x;          // over B*L*E
    float f = values[idx];
    _Float16 hf = (_Float16)f;
    Vh[idx] = hf;
    int e = idx & (E_ - 1);
    int l = (idx >> 9) & (L_ - 1);
    int b = idx >> 21;
    int h = e >> 6;
    int d = e & (D_ - 1);
    Vt[(((size_t)b * H_ + h) * D_ + d) * L_ + l] = hf;
}

// ---------------------------------------------------------------------------
// Prep 2: Ksum[b][l][d] = f16( sum_h values[b][l][h*64+d] )   (B x L x 64)
// ---------------------------------------------------------------------------
__global__ __launch_bounds__(256) void prep_ksum_kernel(
    const float* __restrict__ values, _Float16* __restrict__ Ks) {
    int idx = blockIdx.x * 256 + threadIdx.x;          // over B*L*64
    int d = idx & (D_ - 1);
    int l = (idx >> 6) & (L_ - 1);
    int b = idx >> 18;
    const float* base = values + ((size_t)b * L_ + l) * E_ + d;
    float s = 0.f;
#pragma unroll
    for (int h = 0; h < H_; ++h) s += base[h * D_];
    Ks[idx] = (_Float16)s;
}

// ---------------------------------------------------------------------------
// Prep 3: fc_w (f32 [j][e]) -> Wh (f16)
// ---------------------------------------------------------------------------
__global__ __launch_bounds__(256) void prep_w_kernel(
    const float* __restrict__ w, _Float16* __restrict__ Wh) {
    int idx = blockIdx.x * 256 + threadIdx.x;          // over E*E
    Wh[idx] = (_Float16)w[idx];
}

// ---------------------------------------------------------------------------
// Flash attention: grid (L/64, H, B), block = 128 (4 waves).
// K/V tiles staged in LDS by async DMA, double-buffered, ONE barrier/iter;
// wave w owns q-rows [qt*64 + 16w, +16). Online softmax with DPP reductions.
//
// LDS map (halves, row stride 72 = 144B to spread DS banks):
//   [0      , 4608)  K buffer 0        (64 rows x 72)
//   [4608   , 9216)  K buffer 1
//   [9216   ,13824)  V buffer 0        (64 d-rows x 72)
//   [13824  ,18432)  V buffer 1
//   [18432  ,23040)  P strips, 16x72 per wave
// ---------------------------------------------------------------------------
__global__ __launch_bounds__(128) void attn_kernel(
    const _Float16* __restrict__ Q,   // Vh  [b][l][e]
    const _Float16* __restrict__ K,   // Ks  [b][l][64] (head-summed)
    const _Float16* __restrict__ Vt,  // [b][h][d][l]
    _Float16* __restrict__ O) {       // [b][l][e] (attention out, pre-fc)
    __shared__ __align__(16) _Float16 shm[5 * 4608];

    const int tid  = threadIdx.x;
    const int lane = tid & 31;
    const int wave = tid >> 5;
    const int n16  = lane & 15;
    const int hi   = lane >> 4;
    const int qt = blockIdx.x, h = blockIdx.y, b = blockIdx.z;
    const int qrow0 = qt * 64 + wave * 16;

    const unsigned shmBase = (unsigned)(size_t)(&shm[0]);
    _Float16* pstrip = &shm[4 * 4608 + wave * (16 * 72)];

    const _Float16* kGlob = K + ((size_t)b * L_) * D_;
    const _Float16* vGlob = Vt + (((size_t)b * H_ + h) * D_) * (size_t)L_;

    // Issue one 64x64 K tile + one 64x64 V tile into LDS buffer `buf`.
    // 512 16B-chunks per tile; 4 per thread; 8 async ops per wave total.
    auto issue_tile = [&](int kbase, int buf) {
#pragma unroll
        for (int j = 0; j < 4; ++j) {
            int c   = tid + 128 * j;
            int row = c >> 3;             // 0..63
            int cc  = (c & 7) * 8;        // halves within row (16B chunks)
            unsigned ldsRow = (unsigned)(row * 144 + cc * 2);
            async_copy16(shmBase + buf * 9216 + ldsRow,
                         kGlob + ((size_t)(kbase + row)) * D_ + cc);
            async_copy16(shmBase + 18432 + buf * 9216 + ldsRow,
                         vGlob + (size_t)row * L_ + kbase + cc);
        }
    };

    // --- Q A-fragments (16 rows x 64 d = two 16x32 fragments), kept resident
    issue_tile(0, 0);
    FragA aQ[2];
    const _Float16* qbase = Q + (((size_t)b * L_ + qrow0 + n16) * E_) + h * D_;
#pragma unroll
    for (int s = 0; s < 2; ++s) {
        int d0 = 32 * s + 16 * hi;  // A layout: k = 16*hi + i (contiguous)
        aQ[s].u[0] = *(const u32x4*)(qbase + d0);
        aQ[s].u[1] = *(const u32x4*)(qbase + d0 + 8);
    }

    v8f cO[4] = {};                    // 16x64 O accumulator (4 d-tiles)
    float mr[8], lr[8];
#pragma unroll
    for (int r = 0; r < 8; ++r) { mr[r] = -3.0e38f; lr[r] = 0.f; }

    for (int it = 0; it < L_ / 64; ++it) {
        const int buf = it & 1;
        const _Float16* kbuf = &shm[buf * 4608];
        const _Float16* vbuf = &shm[9216 + buf * 4608];

        S_WAIT_ASYNCCNT(0);   // this wave's chunks of tile `it` landed
        __syncthreads();      // all waves' chunks visible; iter it-1 readers done
        if (it + 1 < L_ / 64)
            issue_tile((it + 1) * 64, buf ^ 1);  // overlaps compute below

        // ---- S = Q x Ksum^T : 4 n-tiles, 2 k-steps each (from LDS) ----
        v8f cS[4];
#pragma unroll
        for (int t = 0; t < 4; ++t) {
            const _Float16* kr = kbuf + (t * 16 + n16) * 72 + 16 * hi;
            FragA bK0, bK1;   // B layout: lane n, k = 16*hi + i (contiguous)
            bK0.u[0] = *(const u32x4*)(kr);
            bK0.u[1] = *(const u32x4*)(kr + 8);
            bK1.u[0] = *(const u32x4*)(kr + 32);
            bK1.u[1] = *(const u32x4*)(kr + 40);
            v8f c = {};
            c = wmma16(aQ[0], bK0, c);
            c = wmma16(aQ[1], bK1, c);
            cS[t] = c;
        }

        // ---- online softmax: row max over 64 cols (tile max + DPP fly) ----
        float alpha[8];
#pragma unroll
        for (int r = 0; r < 8; ++r) {
            float tm = fmaxf(fmaxf(cS[0][r], cS[1][r]),
                             fmaxf(cS[2][r], cS[3][r]));
            tm = row16_max(tm);
            float mn = fmaxf(mr[r], tm * CSC);
            alpha[r] = exp2f(mr[r] - mn);
            mr[r] = mn;
        }

        // ---- P = exp2(S*c - m), write strip to LDS (C -> A layout xpose) ----
        float rs[8];
#pragma unroll
        for (int r = 0; r < 8; ++r) rs[r] = 0.f;
#pragma unroll
        for (int t = 0; t < 4; ++t) {
#pragma unroll
            for (int r = 0; r < 8; ++r) {
                float p = exp2f(cS[t][r] * CSC - mr[r]);
                rs[r] += p;
                pstrip[(r + 8 * hi) * 72 + t * 16 + n16] = (_Float16)p;
            }
        }
#pragma unroll
        for (int r = 0; r < 8; ++r)
            lr[r] = lr[r] * alpha[r] + row16_sum(rs[r]);
#pragma unroll
        for (int t = 0; t < 4; ++t)
#pragma unroll
            for (int r = 0; r < 8; ++r) cO[t][r] *= alpha[r];

        // ---- O += P x V : A from own P strip, B from LDS V tile ----
#pragma unroll
        for (int s = 0; s < 2; ++s) {
            FragA aP;
            const _Float16* pb = pstrip + n16 * 72 + s * 32 + 16 * hi;
            aP.u[0] = *(const u32x4*)(pb);
            aP.u[1] = *(const u32x4*)(pb + 8);
#pragma unroll
            for (int t = 0; t < 4; ++t) {
                const _Float16* vr = vbuf + (t * 16 + n16) * 72 + s * 32 + 16 * hi;
                FragA bV;
                bV.u[0] = *(const u32x4*)(vr);
                bV.u[1] = *(const u32x4*)(vr + 8);
                cO[t] = wmma16(aP, bV, cO[t]);
            }
        }
    }

    // ---- normalize and store O (f16, for the fc WMMA epilogue) ----
    float linv[8];
#pragma unroll
    for (int r = 0; r < 8; ++r) linv[r] = 1.0f / lr[r];
#pragma unroll
    for (int t = 0; t < 4; ++t)
#pragma unroll
        for (int r = 0; r < 8; ++r) {
            float o = cO[t][r] * linv[r];
            O[(((size_t)b * L_ + qrow0 + r + 8 * hi) * E_) +
              h * D_ + t * 16 + n16] = (_Float16)o;
        }
}

// ---------------------------------------------------------------------------
// Epilogue: out[q][j] = sum_e O[q][e] * fc_w[j][e] + fc_b[j]   (f32 out)
// grid (B*L/64, E/64), block 128 (4 waves, 16-row strips)
// ---------------------------------------------------------------------------
__global__ __launch_bounds__(128) void fc_kernel(
    const _Float16* __restrict__ A,   // O f16 [B*L][512]
    const _Float16* __restrict__ W,   // fc_w f16 [j][e]
    const float* __restrict__ bias, float* __restrict__ out) {
    const int lane = threadIdx.x & 31;
    const int wave = threadIdx.x >> 5;
    const int n16 = lane & 15;
    const int hi  = lane >> 4;
    const int row0 = blockIdx.x * 64 + wave * 16;
    const int col0 = blockIdx.y * 64;

    v8f acc[4] = {};
    const _Float16* arow = A + (size_t)(row0 + n16) * E_;
#pragma unroll 4
    for (int k0 = 0; k0 < E_; k0 += 32) {
        FragA a;
        a.u[0] = *(const u32x4*)(arow + k0 + 16 * hi);
        a.u[1] = *(const u32x4*)(arow + k0 + 16 * hi + 8);
#pragma unroll
        for (int t = 0; t < 4; ++t) {
            const _Float16* wrow =
                W + (size_t)(col0 + t * 16 + n16) * E_ + k0 + 16 * hi;
            FragA bw;
            bw.u[0] = *(const u32x4*)(wrow);
            bw.u[1] = *(const u32x4*)(wrow + 8);
            acc[t] = wmma16(a, bw, acc[t]);
        }
    }
#pragma unroll
    for (int t = 0; t < 4; ++t) {
        float bb = bias[col0 + t * 16 + n16];
#pragma unroll
        for (int r = 0; r < 8; ++r)
            out[(size_t)(row0 + r + 8 * hi) * E_ + col0 + t * 16 + n16] =
                acc[t][r] + bb;
    }
}

// ---------------------------------------------------------------------------
extern "C" void kernel_launch(void* const* d_in, const int* in_sizes, int n_in,
                              void* d_out, int out_size, void* d_ws,
                              size_t ws_size, hipStream_t stream) {
    (void)in_sizes; (void)n_in; (void)out_size; (void)ws_size;
    const float* values = (const float*)d_in[0];
    // d_in[1]=keys, d_in[2]=query, d_in[3]=mask are dead per the reference bugs
    const float* fc_w = (const float*)d_in[4];
    const float* fc_b = (const float*)d_in[5];
    float* out = (float*)d_out;

    char* ws = (char*)d_ws;
    _Float16* Vh = (_Float16*)(ws + 0);                 //  8,388,608 B
    _Float16* Ks = (_Float16*)(ws + 8388608);           //  1,048,576 B
    _Float16* Vt = (_Float16*)(ws + 9437184);           //  8,388,608 B
    _Float16* Oh = (_Float16*)(ws + 17825792);          //  8,388,608 B
    _Float16* Wh = (_Float16*)(ws + 26214400);          //    524,288 B

    prep_vals_kernel<<<(B_ * L_ * E_) / 256, 256, 0, stream>>>(values, Vh, Vt);
    prep_ksum_kernel<<<(B_ * L_ * D_) / 256, 256, 0, stream>>>(values, Ks);
    prep_w_kernel<<<(E_ * E_) / 256, 256, 0, stream>>>(fc_w, Wh);

    attn_kernel<<<dim3(L_ / 64, H_, B_), 128, 0, stream>>>(Vh, Ks, Vt, Oh);
    fc_kernel<<<dim3((B_ * L_) / 64, E_ / 64), 128, 0, stream>>>(Oh, Wh, fc_b,
                                                                 out);
}
